// FaceDetectionModel_28664611733777
// MI455X (gfx1250) — matrix operand order repaired and checked by reading, not compiled, not run on hardware
//
#include <hip/hip_runtime.h>
#include <math.h>

// ---------------------------------------------------------------------------
// Types for CDNA5 WMMA (wave32): v_wmma_f32_16x16x32_f16
// ---------------------------------------------------------------------------
typedef __attribute__((ext_vector_type(16))) _Float16 v16h;
typedef __attribute__((ext_vector_type(8)))  float    v8f;
typedef int v4i __attribute__((vector_size(16)));     // builtin's b128 unit

union HU    { _Float16 h; unsigned short u; };
union AFrag { v16h v; unsigned u[8]; uint4 q[2]; };
union CFrag { v8f  v; float f[8]; };

__device__ __forceinline__ float silu_f(float v) { return v / (1.f + __expf(-v)); }
__device__ __forceinline__ _Float16 f2h(float v) { return (_Float16)v; }

// ---------------------------------------------------------------------------
// CDNA5 async global->LDS staging (ASYNCcnt-tracked). Guarded so the file
// still compiles (sync fallback) if the toolchain lacks the builtin.
// Probe-confirmed signature: (v4i addrspace(1)* src, v4i addrspace(3)* dst,
//                             imm int offset, imm int cpol)
// ---------------------------------------------------------------------------
#if defined(__HIP_DEVICE_COMPILE__) && __has_builtin(__builtin_amdgcn_global_load_async_to_lds_b128)
#define HAVE_ASYNC_LDS 1
#else
#define HAVE_ASYNC_LDS 0
#endif

__device__ __forceinline__ void async_g2l_b128(const void* g, void* l)
{
#if HAVE_ASYNC_LDS
    __builtin_amdgcn_global_load_async_to_lds_b128(
        (__attribute__((address_space(1))) v4i*)g,
        (__attribute__((address_space(3))) v4i*)l, 0, 0);
#else
    (void)g; (void)l;
#endif
}

template<int N>
__device__ __forceinline__ void wait_asynccnt()
{
#if defined(__HIP_DEVICE_COMPILE__)
#if __has_builtin(__builtin_amdgcn_s_wait_asynccnt)
    __builtin_amdgcn_s_wait_asynccnt((unsigned short)N);
    asm volatile("" ::: "memory");
#else
    asm volatile("s_wait_asynccnt %0" :: "i"(N) : "memory");
#endif
#endif
}

// ---------------------------------------------------------------------------
// conv1: 3->16 ch, 3x3 stride 2, SAME (pad begin 0 / end 1), SiLU.
// Input NCHW fp32 (16,3,640,640); output NHWC f16 (16,320,320,16).
// K = 27 -> direct conv (memory bound), weights staged in LDS.
// ---------------------------------------------------------------------------
__global__ void __launch_bounds__(256)
conv1_direct(const float* __restrict__ x, const float* __restrict__ W1,
             const float* __restrict__ b1, _Float16* __restrict__ out)
{
    __shared__ float ws[432];   // 16*3*3*3 OIHW
    __shared__ float bs[16];
    for (int i = threadIdx.x; i < 432; i += 256) ws[i] = W1[i];
    if (threadIdx.x < 16) bs[threadIdx.x] = b1[threadIdx.x];
    __syncthreads();

    const int m  = blockIdx.x * 256 + threadIdx.x;   // over 16*320*320 pixels
    const int ox = m % 320;
    int t = m / 320;
    const int oy = t % 320;
    const int b  = t / 320;

    float acc[16];
#pragma unroll
    for (int o = 0; o < 16; ++o) acc[o] = bs[o];

#pragma unroll
    for (int c = 0; c < 3; ++c) {
        const float* xp = x + ((long)b * 3 + c) * 640 * 640;
#pragma unroll
        for (int kh = 0; kh < 3; ++kh) {
            const int iy = oy * 2 + kh;          // pad_begin = 0
            if (iy >= 640) continue;
#pragma unroll
            for (int kw = 0; kw < 3; ++kw) {
                const int ix = ox * 2 + kw;
                if (ix >= 640) continue;
                const float v = xp[(long)iy * 640 + ix];
#pragma unroll
                for (int o = 0; o < 16; ++o)
                    acc[o] += v * ws[(o * 3 + c) * 9 + kh * 3 + kw];
            }
        }
    }

    union { _Float16 h[16]; uint4 q[2]; } pk;
#pragma unroll
    for (int o = 0; o < 16; ++o) pk.h[o] = f2h(silu_f(acc[o]));
    uint4* op = reinterpret_cast<uint4*>(out + (long)m * 16);
    op[0] = pk.q[0];
    op[1] = pk.q[1];
}

// ---------------------------------------------------------------------------
// 3x3 stride-2 conv as implicit GEMM via v_wmma_f32_16x16x32_f16.
//   M = B*OH*OW (16-pixel tile/wave), N = COUT (16-wide tiles),
//   K = 9*CIN in 32-chunks.  K order = (tap, cin), CIN is a power of two, so
//   each lane's A fragment per chunk is exactly two contiguous 16-byte runs:
//     run0: K = chunk*32 + hv*8 .. +7,  run1: K = run0 + 16
//   These are staged global->LDS with async b128 copies (double buffered,
//   ASYNCcnt), then pulled into VGPRs with ds_load_b128.
// B fragment (32x16): lane l -> col n=l%16, rows K=(l/16)*8+e+(e>=8?8:0), pre-swizzled
//   into LDS once per workgroup from fp32 OIHW weights.
// C/D f32: element r -> row M = r + (l/16)*8, col N = l%16; NHWC flat pixel
//   index == GEMM row, so the store address is just mm*COUT + n (no div/mod).
// ---------------------------------------------------------------------------
template<int CIN, int COUT, int HIN, int WIN, int NW>
__global__ void __launch_bounds__(NW * 32)
conv3x3s2_wmma(const _Float16* __restrict__ in, const float* __restrict__ Wt,
               const float* __restrict__ bias, _Float16* __restrict__ out,
               const _Float16* __restrict__ zpad, int tiles)
{
    constexpr int OH = HIN / 2, OW = WIN / 2;
    constexpr int KTOT   = 9 * CIN;
    constexpr int FULL   = KTOT / 32;          // fully populated K chunks
    constexpr int CHUNKS = (KTOT + 31) / 32;   // incl. zero-padded tail
    constexpr int NT     = COUT / 16;
    __shared__ unsigned bsh[CHUNKS * NT * 256];   // [chunk][nt][lane][8] dwords
#if HAVE_ASYNC_LDS
    __shared__ unsigned abuf[NW][2][256];         // per-wave double buffer
#endif

    const int tid = threadIdx.x;

    // Stage B fragments (swizzled f16 weights) into LDS.
    for (int idx = tid; idx < CHUNKS * NT * 256; idx += NW * 32) {
        const int j     = idx & 7;
        const int lane  = (idx >> 3) & 31;
        const int cn    = idx >> 8;               // chunk*NT + nt
        const int nt    = cn % NT;
        const int chunk = cn / NT;
        const int n  = nt * 16 + (lane & 15);
        const int kb = (lane >> 4) * 8;
        unsigned pack = 0;
#pragma unroll
        for (int s = 0; s < 2; ++s) {
            const int K = chunk * 32 + kb + 2 * j + s + ((2 * j + s) >= 8 ? 8 : 0);
            float w = 0.f;
            if (K < KTOT) {
                const int tap = K / CIN, c = K % CIN;
                w = Wt[((n * CIN + c) * 3 + tap / 3) * 3 + (tap % 3)]; // OIHW
            }
            HU cv; cv.h = f2h(w);
            pack |= ((unsigned)cv.u) << (16 * s);
        }
        bsh[idx] = pack;
    }
    __syncthreads();

    const int lane = tid & 31;
    const int wave = tid >> 5;
    const int m15  = lane & 15;
    const int hv   = lane >> 4;
    const int waveStride = gridDim.x * NW;

    for (int tile = blockIdx.x * NW + wave; tile < tiles; tile += waveStride) {
        const int m  = tile * 16 + m15;
        const int b  = m / (OH * OW);            // once per tile only
        const int rm = m % (OH * OW);
        const int oy = rm / OW, ox = rm % OW;

        const _Float16* base = in + (((long)b * HIN + oy * 2) * WIN + ox * 2) * CIN;
        const bool okY = (oy * 2 + 2) < HIN;     // tap row kh==2 in-bounds
        const bool okX = (ox * 2 + 2) < WIN;     // tap col kw==2 in-bounds

        // Global source pointer for one 16-byte A run (8 consecutive K).
        auto runPtr = [&](int K0) -> const _Float16* {
            if (K0 >= KTOT) return zpad;
            const int tap = K0 / CIN, c0 = K0 % CIN;      // CIN pow2 -> shifts
            const int kh = tap / 3, kw = tap % 3;
            const bool ok = (kh < 2 || okY) && (kw < 2 || okX);
            return ok ? (base + ((kh * WIN + kw) * CIN + c0)) : zpad;
        };

        CFrag acc[NT];
#pragma unroll
        for (int nt = 0; nt < NT; ++nt) {
            const float bv = bias[nt * 16 + m15];
#pragma unroll
            for (int r = 0; r < 8; ++r) acc[nt].f[r] = bv;
        }

        auto doWmma = [&](const AFrag& a, int chunk) {
#pragma unroll
            for (int nt = 0; nt < NT; ++nt) {
                AFrag bf;
                const uint4* bp =
                    reinterpret_cast<const uint4*>(&bsh[(chunk * NT + nt) * 256 + lane * 8]);
                bf.q[0] = bp[0];
                bf.q[1] = bp[1];
                acc[nt].v = __builtin_amdgcn_wmma_f32_16x16x32_f16(
                    false, a.v, false, bf.v, (short)0, acc[nt].v, false, false);
            }
        };

        // Sync loader (fallback + zero-padded tail chunk).
        auto loadSync = [&](int chunk) -> AFrag {
            AFrag a;
            const int K0 = chunk * 32 + hv * 8;
            const _Float16* g0 = runPtr(K0);
            const _Float16* g1 = runPtr(K0 + 16);
            a.q[0] = *reinterpret_cast<const uint4*>(g0);
            a.q[1] = *reinterpret_cast<const uint4*>(g1);
            return a;
        };

#if HAVE_ASYNC_LDS
        auto issueChunk = [&](int chunk, int pb) {
            const int K0 = chunk * 32 + hv * 8;
            async_g2l_b128(runPtr(K0),      &abuf[wave][pb][lane * 8]);
            async_g2l_b128(runPtr(K0 + 16), &abuf[wave][pb][lane * 8 + 4]);
        };

        issueChunk(0, 0);
#pragma unroll
        for (int c = 0; c < FULL; ++c) {
            if (c + 1 < FULL) { issueChunk(c + 1, (c + 1) & 1); wait_asynccnt<2>(); }
            else              { wait_asynccnt<0>(); }
            AFrag a;
            const uint4* ar =
                reinterpret_cast<const uint4*>(&abuf[wave][c & 1][lane * 8]);
            a.q[0] = ar[0];
            a.q[1] = ar[1];
            doWmma(a, c);
        }
#else
#pragma unroll
        for (int c = 0; c < FULL; ++c) {
            AFrag a = loadSync(c);
            doWmma(a, c);
        }
#endif
        if constexpr (CHUNKS > FULL) {          // zero-padded K tail
            AFrag a = loadSync(FULL);
            doWmma(a, FULL);
        }

        // Fused SiLU epilogue; NHWC flat pixel == GEMM row -> address mm*COUT+n.
#pragma unroll
        for (int nt = 0; nt < NT; ++nt) {
#pragma unroll
            for (int r = 0; r < 8; ++r) {
                const long mm = (long)tile * 16 + r + hv * 8;
                out[mm * COUT + nt * 16 + m15] = f2h(silu_f(acc[nt].f[r]));
            }
        }
    }
}

// ---------------------------------------------------------------------------
// 1x1 head conv 64->48, WMMA, f32 output (no activation; decode sigmoids).
// A runs are fully contiguous per pixel; async-staged like the 3x3 convs.
// ---------------------------------------------------------------------------
template<int NW>
__global__ void __launch_bounds__(NW * 32)
head1x1_wmma(const _Float16* __restrict__ in, const float* __restrict__ Wt,
             const float* __restrict__ bias, float* __restrict__ out, int tiles)
{
    constexpr int CIN = 64, COUT = 48, FULL = 2, NT = 3;
    __shared__ unsigned bsh[FULL * NT * 256];
#if HAVE_ASYNC_LDS
    __shared__ unsigned abuf[NW][2][256];
#endif

    const int tid = threadIdx.x;
    for (int idx = tid; idx < FULL * NT * 256; idx += NW * 32) {
        const int j     = idx & 7;
        const int lane  = (idx >> 3) & 31;
        const int cn    = idx >> 8;
        const int nt    = cn % NT;
        const int chunk = cn / NT;
        const int n  = nt * 16 + (lane & 15);
        const int kb = (lane >> 4) * 8;
        unsigned pack = 0;
#pragma unroll
        for (int s = 0; s < 2; ++s) {
            const int K = chunk * 32 + kb + 2 * j + s + ((2 * j + s) >= 8 ? 8 : 0);
            HU cv; cv.h = f2h(Wt[n * CIN + K]);   // Wh is (48,64,1,1)
            pack |= ((unsigned)cv.u) << (16 * s);
        }
        bsh[idx] = pack;
    }
    __syncthreads();

    const int lane = tid & 31;
    const int wave = tid >> 5;
    const int m15  = lane & 15;
    const int hv   = lane >> 4;
    const int waveStride = gridDim.x * NW;

    for (int tile = blockIdx.x * NW + wave; tile < tiles; tile += waveStride) {
        const int m = tile * 16 + m15;
        const _Float16* base = in + (long)m * CIN;

        CFrag acc[NT];
#pragma unroll
        for (int nt = 0; nt < NT; ++nt) {
            const float bv = bias[nt * 16 + m15];
#pragma unroll
            for (int r = 0; r < 8; ++r) acc[nt].f[r] = bv;
        }

        auto doWmma = [&](const AFrag& a, int chunk) {
#pragma unroll
            for (int nt = 0; nt < NT; ++nt) {
                AFrag bf;
                const uint4* bp =
                    reinterpret_cast<const uint4*>(&bsh[(chunk * NT + nt) * 256 + lane * 8]);
                bf.q[0] = bp[0];
                bf.q[1] = bp[1];
                acc[nt].v = __builtin_amdgcn_wmma_f32_16x16x32_f16(
                    false, a.v, false, bf.v, (short)0, acc[nt].v, false, false);
            }
        };

#if HAVE_ASYNC_LDS
        auto issueChunk = [&](int chunk, int pb) {
            const int K0 = chunk * 32 + hv * 8;
            async_g2l_b128(base + K0,      &abuf[wave][pb][lane * 8]);
            async_g2l_b128(base + K0 + 16, &abuf[wave][pb][lane * 8 + 4]);
        };
        issueChunk(0, 0);
#pragma unroll
        for (int c = 0; c < FULL; ++c) {
            if (c + 1 < FULL) { issueChunk(c + 1, (c + 1) & 1); wait_asynccnt<2>(); }
            else              { wait_asynccnt<0>(); }
            AFrag a;
            const uint4* ar =
                reinterpret_cast<const uint4*>(&abuf[wave][c & 1][lane * 8]);
            a.q[0] = ar[0];
            a.q[1] = ar[1];
            doWmma(a, c);
        }
#else
#pragma unroll
        for (int c = 0; c < FULL; ++c) {
            AFrag a;
            const int K0 = c * 32 + hv * 8;
            a.q[0] = *reinterpret_cast<const uint4*>(base + K0);
            a.q[1] = *reinterpret_cast<const uint4*>(base + K0 + 16);
            doWmma(a, c);
        }
#endif

#pragma unroll
        for (int nt = 0; nt < NT; ++nt)
#pragma unroll
            for (int r = 0; r < 8; ++r) {
                const long mm = (long)tile * 16 + r + hv * 8;
                out[mm * COUT + nt * 16 + m15] = acc[nt].f[r];
            }
    }
}

// ---------------------------------------------------------------------------
// Decode: sigmoid + box decode + conf threshold.
// h: [B,80,80,48] f32 ; boxes: [B,19200,4] ; scores: [B,19200]
// flat index within batch = (a*80 + gy)*80 + gx  (matches reshape(B,-1))
// ---------------------------------------------------------------------------
__global__ void __launch_bounds__(256)
decode_kernel(const float* __restrict__ h, float* __restrict__ boxes,
              float* __restrict__ scores)
{
    const int i  = blockIdx.x * 256 + threadIdx.x;   // B*3*80*80 = 307200 exact
    const int gx = i % 80;
    int t = i / 80;
    const int gy = t % 80; t /= 80;
    const int a  = t % 3;
    const int b  = t / 3;

    const float anchw[3] = {4.f, 8.f, 13.f};
    const float anchh[3] = {5.f, 10.f, 16.f};

    const float* hp = h + ((long)(b * 80 + gy) * 80 + gx) * 48 + a * 16;
    float p[16];
#pragma unroll
    for (int c = 0; c < 16; ++c) p[c] = 1.f / (1.f + __expf(-hp[c]));

    const float cx = (p[0] * 2.f - 0.5f + (float)gx) * 8.f;
    const float cy = (p[1] * 2.f - 0.5f + (float)gy) * 8.f;
    const float w  = (p[2] * 2.f) * (p[2] * 2.f) * anchw[a];
    const float hh = (p[3] * 2.f) * (p[3] * 2.f) * anchh[a];
    float sc = p[4] * p[15];
    sc = sc > 0.5f ? sc : 0.f;

    const int oi = (a * 80 + gy) * 80 + gx;
    float* bb = boxes + ((long)b * 19200 + oi) * 4;
    bb[0] = cx - w * 0.5f;
    bb[1] = cy - hh * 0.5f;
    bb[2] = cx + w * 0.5f;
    bb[3] = cy + hh * 0.5f;
    scores[(long)b * 19200 + oi] = sc;
}

// ---------------------------------------------------------------------------
// Top-300 selection (argmax-with-removal, mutates scores in workspace) +
// greedy NMS + output packing. One workgroup (512 thr = 16 waves) per batch.
// ---------------------------------------------------------------------------
__global__ void __launch_bounds__(512)
nms_kernel(const float* __restrict__ boxes, float* __restrict__ scores,
           float* __restrict__ out)
{
    const int b   = blockIdx.x;
    const int tid = threadIdx.x;
    __shared__ float selb[300 * 4];
    __shared__ float sels[300];
    __shared__ int   keep[300];
    __shared__ float rv[16];
    __shared__ int   ri[16];
    __shared__ int   winner;

    float*       sc = scores + (long)b * 19200;
    const float* bx = boxes  + (long)b * 19200 * 4;

    for (int it = 0; it < 300; ++it) {
        float bv = -1e30f;
        int   bi = 1 << 30;
        for (int i = tid; i < 19200; i += 512) {
            const float v = sc[i];
            if (v > bv || (v == bv && i < bi)) { bv = v; bi = i; }
        }
        for (int o = 16; o; o >>= 1) {
            const float ov = __shfl_down(bv, o, 32);
            const int   oi = __shfl_down(bi, o, 32);
            if (ov > bv || (ov == bv && oi < bi)) { bv = ov; bi = oi; }
        }
        if ((tid & 31) == 0) { rv[tid >> 5] = bv; ri[tid >> 5] = bi; }
        __syncthreads();
        if (tid == 0) {
            float mv = rv[0]; int mi = ri[0];
            for (int w = 1; w < 16; ++w)
                if (rv[w] > mv || (rv[w] == mv && ri[w] < mi)) { mv = rv[w]; mi = ri[w]; }
            sels[it] = mv;
            winner   = mi;
            sc[mi]   = -1e30f;   // remove
            __threadfence();
        }
        __syncthreads();
        if (tid < 4) selb[it * 4 + tid] = bx[(long)winner * 4 + tid];
        __syncthreads();
    }

    for (int i = tid; i < 300; i += 512) keep[i] = 1;
    __syncthreads();

    for (int i = 0; i < 300; ++i) {
        if (keep[i] && tid < 300 && tid > i && keep[tid]) {
            const float x1a = selb[i * 4 + 0], y1a = selb[i * 4 + 1];
            const float x2a = selb[i * 4 + 2], y2a = selb[i * 4 + 3];
            const float x1b = selb[tid * 4 + 0], y1b = selb[tid * 4 + 1];
            const float x2b = selb[tid * 4 + 2], y2b = selb[tid * 4 + 3];
            const float areaA = fmaxf(x2a - x1a, 0.f) * fmaxf(y2a - y1a, 0.f);
            const float areaB = fmaxf(x2b - x1b, 0.f) * fmaxf(y2b - y1b, 0.f);
            const float ix1 = fmaxf(x1a, x1b), iy1 = fmaxf(y1a, y1b);
            const float ix2 = fminf(x2a, x2b), iy2 = fminf(y2a, y2b);
            const float inter = fmaxf(ix2 - ix1, 0.f) * fmaxf(iy2 - iy1, 0.f);
            const float iou   = inter / (areaA + areaB - inter + 1e-7f);
            if (iou > 0.5f) keep[tid] = 0;
        }
        __syncthreads();
    }

    for (int i = tid; i < 300; i += 512) {
        float* o = out + ((long)b * 300 + i) * 6;
        o[0] = selb[i * 4 + 0];
        o[1] = selb[i * 4 + 1];
        o[2] = selb[i * 4 + 2];
        o[3] = selb[i * 4 + 3];
        o[4] = sels[i];
        o[5] = (keep[i] && sels[i] > 0.f) ? 1.f : 0.f;
    }
}

// ---------------------------------------------------------------------------
// Host launcher
// ---------------------------------------------------------------------------
extern "C" void kernel_launch(void* const* d_in, const int* in_sizes, int n_in,
                              void* d_out, int out_size, void* d_ws, size_t ws_size,
                              hipStream_t stream)
{
    (void)in_sizes; (void)n_in; (void)out_size; (void)ws_size;
    const float* x  = (const float*)d_in[0];
    const float* W1 = (const float*)d_in[1];
    const float* b1 = (const float*)d_in[2];
    const float* W2 = (const float*)d_in[3];
    const float* b2 = (const float*)d_in[4];
    const float* W3 = (const float*)d_in[5];
    const float* b3 = (const float*)d_in[6];
    const float* Wh = (const float*)d_in[7];
    const float* bh = (const float*)d_in[8];
    float* out = (float*)d_out;

    // Workspace layout (bytes), ~118 MB total:
    char* ws = (char*)d_ws;
    _Float16* f1   = (_Float16*)(ws);                 // 16*320*320*16 f16 = 52,428,800
    _Float16* f2   = (_Float16*)(ws + 52428800);      // 16*160*160*32 f16 = 26,214,400
    _Float16* f3   = (_Float16*)(ws + 78643200);      // 16*80*80*64  f16 = 13,107,200
    float*    hd   = (float*)   (ws + 91750400);      // 16*80*80*48  f32 = 19,660,800
    float*    bxs  = (float*)   (ws + 111411200);     // 16*19200*4   f32 =  4,915,200
    float*    scs  = (float*)   (ws + 116326400);     // 16*19200     f32 =  1,228,800
    _Float16* zpad = (_Float16*)(ws + 117555200);     // 256 B zero pad for OOB taps

    // Zero the OOB pad (capture-legal stream op).
    (void)hipMemsetAsync((void*)zpad, 0, 256, stream);

    // conv1: 16*320*320 = 1,638,400 pixels
    conv1_direct<<<6400, 256, 0, stream>>>(x, W1, b1, f1);

    // conv2: M = 16*160*160 = 409600 -> 25600 wave-tiles
    conv3x3s2_wmma<16, 32, 320, 320, 8><<<1600, 256, 0, stream>>>(f1, W2, b2, f2, zpad, 25600);

    // conv3: M = 16*80*80 = 102400 -> 6400 wave-tiles
    conv3x3s2_wmma<32, 64, 160, 160, 8><<<800, 256, 0, stream>>>(f2, W3, b3, f3, zpad, 6400);

    // head: M = 102400 -> 6400 wave-tiles
    head1x1_wmma<8><<<800, 256, 0, stream>>>(f3, Wh, bh, hd, 6400);

    // decode: 16*3*80*80 = 307200 items
    decode_kernel<<<1200, 256, 0, stream>>>(hd, bxs, scs);

    // top-K + NMS: one workgroup per batch
    nms_kernel<<<16, 512, 0, stream>>>(bxs, scs, out);
}
